// ResNeXtBlock_73418170958018
// MI455X (gfx1250) — matrix operand order repaired and checked by reading, not compiled, not run on hardware
//
#include <hip/hip_runtime.h>

#define NPATH 4
#define KOFF 27
#define CIN 256
#define NIN 64
#define EPSV 1e-5f

typedef __attribute__((ext_vector_type(16))) __bf16        bf16x16;
typedef __attribute__((ext_vector_type(8)))  float         floatx8;
typedef __attribute__((ext_vector_type(8)))  unsigned int  u32x8;

__device__ __forceinline__ floatx8 wmma_bf16(bf16x16 a, bf16x16 b, floatx8 c) {
    return __builtin_amdgcn_wmma_f32_16x16x32_bf16(
        false, a, false, b, (short)0, c, false, false);
}

__device__ __forceinline__ void atomAddF(float* p, float v) {
    __hip_atomic_fetch_add(p, v, __ATOMIC_RELAXED, __HIP_MEMORY_SCOPE_AGENT);
}

// A-matrix (16x32 bf16): lane half, element t -> K index
__device__ __forceinline__ int a_kidx(int t, int half) {
    return t + 8 * half + (t >= 8 ? 8 : 0);
}
// B-matrix (32x16 bf16): lane half, element t -> K index
__device__ __forceinline__ int b_kidx(int t, int half) {
    return t + 8 * half + (t >= 8 ? 8 : 0);
}

__device__ __forceinline__ void split_store(bf16x16& hi, bf16x16& lo, int t, float v) {
    __bf16 vh = (__bf16)v;
    hi[t] = vh;
    lo[t] = (__bf16)(v - (float)vh);
}

// B fragment from chunk-major lane-interleaved pack: [tile][chunk 0..3][lane][16B]
// chunks 0,1 = hi bf16 (t 0..7 / 8..15), chunks 2,3 = lo bf16. 4x perfectly coalesced b128.
__device__ __forceinline__ void load_b4(const uint4* q /* tile base: 128 uint4 */, int lane,
                                        bf16x16& bh, bf16x16& blo) {
    uint4 c0 = q[0 * 32 + lane];
    uint4 c1 = q[1 * 32 + lane];
    uint4 c2 = q[2 * 32 + lane];
    uint4 c3 = q[3 * 32 + lane];
    u32x8 hi, lo;
    hi[0] = c0.x; hi[1] = c0.y; hi[2] = c0.z; hi[3] = c0.w;
    hi[4] = c1.x; hi[5] = c1.y; hi[6] = c1.z; hi[7] = c1.w;
    lo[0] = c2.x; lo[1] = c2.y; lo[2] = c2.z; lo[3] = c2.w;
    lo[4] = c3.x; lo[5] = c3.y; lo[6] = c3.z; lo[7] = c3.w;
    bh  = __builtin_bit_cast(bf16x16, hi);
    blo = __builtin_bit_cast(bf16x16, lo);
}

// A fragment from a bf16 plane: per lane two contiguous 16B chunks -> 2x b128
__device__ __forceinline__ bf16x16 load_a_plane(const unsigned short* __restrict__ plane,
                                                int half) {
    const uint4* pp = (const uint4*)plane;     // 64B = whole 32-channel chunk
    uint4 q0 = pp[half];                       // k = 8*half .. +7
    uint4 q1 = pp[2 + half];                   // k = 16+8*half .. +7
    u32x8 a;
    a[0] = q0.x; a[1] = q0.y; a[2] = q0.z; a[3] = q0.w;
    a[4] = q1.x; a[5] = q1.y; a[6] = q1.z; a[7] = q1.w;
    return __builtin_bit_cast(bf16x16, a);
}

// 16 f32 for an A fragment from a row-major f32 row segment: 4x b128, av[t] == frag elem t
__device__ __forceinline__ void load_a_f32(const float* __restrict__ rowseg, int half,
                                           float* av) {
    const float4* xp = (const float4*)rowseg;
    *(float4*)&av[0]  = xp[2 * half];
    *(float4*)&av[4]  = xp[2 * half + 1];
    *(float4*)&av[8]  = xp[4 + 2 * half];
    *(float4*)&av[12] = xp[5 + 2 * half];
}

// ---------------------------------------------------------------- init ----
__global__ void init_ws_kernel(float* __restrict__ zbase, size_t zcount,
                               float* __restrict__ hrow, int hcount) {
    size_t i = (size_t)blockIdx.x * blockDim.x + threadIdx.x;
    if (i < zcount) zbase[i] = 0.0f;
    else if (i < zcount + (size_t)hcount) hrow[i - zcount] = 0.0f;
}

// ------------------------------------------- pre-pass: pack B weights ----
// W: [batch][K][Ncols] f32 -> bf16 hi/lo, chunk-major lane-interleaved frag layout.
__global__ void pack_b_kernel(const float* __restrict__ W, unsigned short* __restrict__ out,
                              int K, int Ncols, int batch) {
    int idx  = blockIdx.x * 256 + threadIdx.x;
    int TPB  = (K / 32) * (Ncols / 16);
    int t    = idx & 15;
    int lane = (idx >> 4) & 31;
    int tile = idx >> 9;
    if (tile >= batch * TPB) return;
    int b   = tile / TPB;
    int rem = tile % TPB;
    int kc  = rem / (Ncols / 16);
    int ct  = rem % (Ncols / 16);
    int half = lane >> 4, n = lane & 15;
    int k = kc * 32 + b_kidx(t, half);
    float v = W[((size_t)b * K + k) * Ncols + ct * 16 + n];
    __bf16 vh = (__bf16)v;
    __bf16 vl = (__bf16)(v - (float)vh);
    size_t tbase = (size_t)tile * 1024;          // u16 per tile
    int c = t >> 3, e = t & 7;
    out[tbase + ((size_t)(c)     * 32 + lane) * 8 + e] = __builtin_bit_cast(unsigned short, vh);
    out[tbase + ((size_t)(2 + c) * 32 + lane) * 8 + e] = __builtin_bit_cast(unsigned short, vl);
}

// ----------------------------------------------------- stage A: linear1 ----
__global__ __launch_bounds__(128) void linear1_kernel(
    const float* __restrict__ x, const unsigned short* __restrict__ WlPack,
    const float* __restrict__ bl, unsigned short* __restrict__ h_pack, int n) {
    const int lane = threadIdx.x & 31;
    const int p    = threadIdx.x >> 5;   // wave -> path
    const int m    = lane & 15;
    const int half = lane >> 4;
    const int row0 = blockIdx.x * 16;
    const int arow = row0 + m;
    const int arowc = arow < n ? arow : n - 1;   // clamp: OOB rows never stored
    const bool full = (row0 + 16 <= n);

    floatx8 accv[4] = {};
    for (int kc = 0; kc < 8; ++kc) {
        float av[16];
        load_a_f32(x + (size_t)arowc * CIN + kc * 32, half, av);
        bf16x16 ah, al;
#pragma unroll
        for (int t = 0; t < 16; ++t) split_store(ah, al, t, av[t]);
#pragma unroll
        for (int ct = 0; ct < 4; ++ct) {
            bf16x16 bh, blo;
            load_b4((const uint4*)WlPack + (size_t)((p * 8 + kc) * 4 + ct) * 128, lane, bh, blo);
            accv[ct] = wmma_bf16(al, bh,  accv[ct]);
            accv[ct] = wmma_bf16(ah, blo, accv[ct]);
            accv[ct] = wmma_bf16(ah, bh,  accv[ct]);
        }
    }
#pragma unroll
    for (int ct = 0; ct < 4; ++ct) {
        const int col  = p * 64 + ct * 16 + (lane & 15);
        const float bi = bl[col];
#pragma unroll
        for (int r = 0; r < 8; ++r) {
            int row = row0 + r + 8 * half;
            if (full || row < n) {
                float v = accv[ct][r] + bi;
                __bf16 vh = (__bf16)v;
                __bf16 vl = (__bf16)(v - (float)vh);
                unsigned short* hr = h_pack + (size_t)row * 512;
                hr[col]       = __builtin_bit_cast(unsigned short, vh);
                hr[256 + col] = __builtin_bit_cast(unsigned short, vl);
            }
        }
    }
}

// -------------------------------------- stage B: gather-GEMM-scatter ----
// Block: one (p,k), 8 waves = 8 m-tiles; Wc fragments staged in LDS (16 KB).
__global__ __launch_bounds__(256) void conv_kernel(
    const unsigned short* __restrict__ h_pack, const unsigned short* __restrict__ WcPack,
    const int* __restrict__ in_idx, const int* __restrict__ out_idx,
    float* __restrict__ acc, int n, int max_m) {
    __shared__ uint4 sWc[1024];                       // 8 tiles * 128 uint4 = 16 KB
    const int pk = blockIdx.y;
    {
        const uint4* src = (const uint4*)WcPack + (size_t)pk * 1024;
#pragma unroll
        for (int i = 0; i < 4; ++i)
            sWc[threadIdx.x + 256 * i] = src[threadIdx.x + 256 * i];
    }
    __syncthreads();

    const int lane = threadIdx.x & 31;
    const int wv   = threadIdx.x >> 5;
    const int m    = lane & 15;
    const int half = lane >> 4;
    const int p    = pk / KOFF;
    const int m0   = (blockIdx.x * 8 + wv) * 16;
    const bool full = (m0 + 16 <= max_m);

    const int* iin  = in_idx  + (size_t)pk * max_m;
    const int* iout = out_idx + (size_t)pk * max_m;
    __builtin_prefetch(iout + m0, 0, 0);

    int arow;
    if (full) arow = iin[m0 + m];
    else      arow = (m0 + m < max_m) ? iin[m0 + m] : n;   // n -> zero sentinel row

    const unsigned short* hrow = h_pack + (size_t)arow * 512 + p * 64;

    floatx8 accv[4] = {};
#pragma unroll
    for (int kc = 0; kc < 2; ++kc) {
        bf16x16 ah = load_a_plane(hrow + kc * 32,       half);
        bf16x16 al = load_a_plane(hrow + 256 + kc * 32, half);
#pragma unroll
        for (int ct = 0; ct < 4; ++ct) {
            bf16x16 bh, blo;
            load_b4(&sWc[(kc * 4 + ct) * 128], lane, bh, blo);   // ds_load_b128 x4
            accv[ct] = wmma_bf16(al, bh,  accv[ct]);
            accv[ct] = wmma_bf16(ah, blo, accv[ct]);
            accv[ct] = wmma_bf16(ah, bh,  accv[ct]);
        }
    }

    float* accb = acc + (size_t)p * (n + 1) * 64 + (lane & 15);
    if (full) {
        const int4* io = (const int4*)(iout + m0 + 8 * half);
        int4 o0 = io[0], o1 = io[1];
        int orow[8] = {o0.x, o0.y, o0.z, o0.w, o1.x, o1.y, o1.z, o1.w};
#pragma unroll
        for (int ct = 0; ct < 4; ++ct)
#pragma unroll
            for (int r = 0; r < 8; ++r)
                atomAddF(accb + (size_t)orow[r] * 64 + ct * 16, accv[ct][r]);
    } else {
#pragma unroll
        for (int ct = 0; ct < 4; ++ct)
#pragma unroll
            for (int r = 0; r < 8; ++r) {
                int mo = m0 + r + 8 * half;
                if (mo < max_m)
                    atomAddF(accb + (size_t)iout[mo] * 64 + ct * 16, accv[ct][r]);
            }
    }
}

// -------------------------------------------------- stage C: BN stats ----
__global__ __launch_bounds__(256) void bn_stats_kernel(
    const float* __restrict__ acc, float* __restrict__ stats, int n) {
    const int p  = blockIdx.y;
    const int ch = threadIdx.x & 63;
    const int rg = threadIdx.x >> 6;
    const float* base = acc + (size_t)p * (n + 1) * 64;
    float s1 = 0.0f, s2 = 0.0f;
    for (int row = blockIdx.x * 4 + rg; row < n; row += gridDim.x * 4) {
        float v = base[(size_t)row * 64 + ch];
        s1 += v; s2 += v * v;
    }
    __shared__ float sh1[256], sh2[256];
    sh1[threadIdx.x] = s1; sh2[threadIdx.x] = s2;
    __syncthreads();
    if (rg == 0) {
        float t1 = sh1[ch] + sh1[64 + ch] + sh1[128 + ch] + sh1[192 + ch];
        float t2 = sh2[ch] + sh2[64 + ch] + sh2[128 + ch] + sh2[192 + ch];
        atomAddF(&stats[p * 128 + ch], t1);
        atomAddF(&stats[p * 128 + 64 + ch], t2);
    }
}

__global__ void bn_finalize_kernel(const float* __restrict__ stats,
                                   const float* __restrict__ gamma,
                                   const float* __restrict__ beta,
                                   float* __restrict__ scl, float* __restrict__ sft,
                                   int n) {
    int i = threadIdx.x;             // 256 = P*64
    int p = i >> 6, ch = i & 63;
    float inv = 1.0f / (float)n;
    float mu  = stats[p * 128 + ch] * inv;
    float var = stats[p * 128 + 64 + ch] * inv - mu * mu;
    float s   = gamma[i] * rsqrtf(var + EPSV);
    scl[i] = s;
    sft[i] = beta[i] - mu * s;
}

// ------------------------- stage D: BN-apply + ReLU + GEMM + residual ----
__global__ __launch_bounds__(128) void final_kernel(
    const float* __restrict__ acc, const float* __restrict__ scl_g,
    const float* __restrict__ sft_g, const unsigned short* __restrict__ W2Pack,
    const float* __restrict__ b2, const float* __restrict__ x,
    float* __restrict__ out, int n) {
    __shared__ float s_scl[256], s_sft[256];
    for (int i = threadIdx.x; i < 256; i += 128) { s_scl[i] = scl_g[i]; s_sft[i] = sft_g[i]; }
    __syncthreads();

    const int lane = threadIdx.x & 31;
    const int w    = threadIdx.x >> 5;   // wave -> 64-col group
    const int m    = lane & 15;
    const int half = lane >> 4;
    const int row0 = blockIdx.x * 16;
    const int arow = row0 + m;
    const int arowc = arow < n ? arow : n - 1;
    const bool full = (row0 + 16 <= n);

    floatx8 accv[4] = {};
    for (int kc = 0; kc < 8; ++kc) {
        const int kbase = kc * 32;
        const int p   = kbase >> 6;        // path constant within a 32-chunk
        const int ch0 = kbase & 63;
        float av[16];
        load_a_f32(acc + ((size_t)p * (n + 1) + arowc) * 64 + ch0, half, av);
        bf16x16 ah, al;
#pragma unroll
        for (int t = 0; t < 16; ++t) {
            int kg = kbase + a_kidx(t, half);
            float v = fmaxf(av[t] * s_scl[kg] + s_sft[kg], 0.0f);  // BN + ReLU fused
            split_store(ah, al, t, v);
        }
#pragma unroll
        for (int ct = 0; ct < 4; ++ct) {
            bf16x16 bh, blo;
            load_b4((const uint4*)W2Pack + (size_t)(kc * 16 + (w * 4 + ct)) * 128, lane, bh, blo);
            accv[ct] = wmma_bf16(al, bh,  accv[ct]);
            accv[ct] = wmma_bf16(ah, blo, accv[ct]);
            accv[ct] = wmma_bf16(ah, bh,  accv[ct]);
        }
    }
#pragma unroll
    for (int ct = 0; ct < 4; ++ct) {
        const int col  = w * 64 + ct * 16 + (lane & 15);
        const float bias = b2[col];
#pragma unroll
        for (int r = 0; r < 8; ++r) {
            int row = row0 + r + 8 * half;
            if (full || row < n)
                out[(size_t)row * 256 + col] = accv[ct][r] + bias + x[(size_t)row * 256 + col];
        }
    }
}

// -------------------------------------------------------------- launch ----
extern "C" void kernel_launch(void* const* d_in, const int* in_sizes, int n_in,
                              void* d_out, int out_size, void* d_ws, size_t ws_size,
                              hipStream_t stream) {
    const float* x     = (const float*)d_in[0];
    const float* Wl    = (const float*)d_in[1];
    const float* bl    = (const float*)d_in[2];
    const float* Wc    = (const float*)d_in[3];
    const float* gamma = (const float*)d_in[4];
    const float* beta  = (const float*)d_in[5];
    const float* W2    = (const float*)d_in[6];
    const float* b2    = (const float*)d_in[7];
    const int*   in_idx  = (const int*)d_in[8];
    const int*   out_idx = (const int*)d_in[9];
    float* out = (float*)d_out;

    const int n     = in_sizes[0] / CIN;              // 200000
    const int max_m = in_sizes[8] / (NPATH * KOFF);

    // ---- workspace layout ----
    char* base = (char*)d_ws;
    unsigned short* h_pack = (unsigned short*)base;                  // (n+1)*512 u16
    size_t hbytes = (size_t)(n + 1) * 512 * sizeof(unsigned short);
    float* acc = (float*)(base + hbytes);                            // 4*(n+1)*64 f32
    size_t Asz = (size_t)NPATH * (n + 1) * 64;
    float* stats = acc + Asz;                                        // 512 f32
    float* scl   = stats + 2 * NPATH * 64;                           // 256 f32
    float* sft   = scl + NPATH * 64;                                 // 256 f32
    unsigned short* WlPack = (unsigned short*)(sft + NPATH * 64);    // 32 tiles * 1024 u16 * 4
    size_t wl_u16 = (size_t)NPATH * 32 * 1024;
    unsigned short* WcPack = WlPack + wl_u16;                        // 108*8 tiles * 1024 u16
    size_t wc_u16 = (size_t)NPATH * KOFF * 8 * 1024;
    unsigned short* W2Pack = WcPack + wc_u16;                        // 128 tiles * 1024 u16

    // zero acc+stats (contiguous) and the sentinel row of h_pack (512 u16 = 256 f32)
    size_t zcount = Asz + 2 * NPATH * 64;
    size_t total  = zcount + 256;
    init_ws_kernel<<<(unsigned)((total + 255) / 256), 256, 0, stream>>>(
        acc, zcount, (float*)(h_pack + (size_t)n * 512), 256);

    // pack weights into WMMA B-fragment layout (bf16 hi/lo, chunk-major)
    pack_b_kernel<<<(NPATH * 32 * 512) / 256, 256, 0, stream>>>(Wl, WlPack, CIN, NIN, NPATH);
    pack_b_kernel<<<(NPATH * KOFF * 8 * 512) / 256, 256, 0, stream>>>(Wc, WcPack, NIN, NIN, NPATH * KOFF);
    pack_b_kernel<<<(128 * 512) / 256, 256, 0, stream>>>(W2, W2Pack, 256, 256, 1);

    const int rtiles = (n + 15) / 16;
    linear1_kernel<<<rtiles, 128, 0, stream>>>(x, WlPack, bl, h_pack, n);

    dim3 cgrid((max_m + 127) / 128, NPATH * KOFF);
    conv_kernel<<<cgrid, 256, 0, stream>>>(h_pack, WcPack, in_idx, out_idx, acc, n, max_m);

    dim3 sgrid(512, NPATH);
    bn_stats_kernel<<<sgrid, 256, 0, stream>>>(acc, stats, n);
    bn_finalize_kernel<<<1, 256, 0, stream>>>(stats, gamma, beta, scl, sft, n);

    final_kernel<<<rtiles, 128, 0, stream>>>(acc, scl, sft, W2Pack, b2, x, out, n);

    (void)n_in; (void)out_size; (void)ws_size;
}
